// RecurrentLayer_53584011985391
// MI455X (gfx1250) — compile-verified, matrix operand order checked by reading, not run on hardware
//
#include <hip/hip_runtime.h>

#define SEQ_    512
#define BATCH_  128
#define INPUT_  128
#define HIDDEN_ 512

typedef _Float16 f16x8 __attribute__((ext_vector_type(8)));
typedef _Float16 v16h  __attribute__((ext_vector_type(16)));
typedef float    v8f   __attribute__((ext_vector_type(8)));

// ---------- prologue helpers ----------
__global__ void cvt_f32_to_f16(const float* __restrict__ src,
                               _Float16* __restrict__ dst, int n) {
  int i = blockIdx.x * blockDim.x + threadIdx.x;
  if (i < n) dst[i] = (_Float16)src[i];
}

__global__ void zero_f16(_Float16* __restrict__ dst, int n) {
  int i = blockIdx.x * blockDim.x + threadIdx.x;
  if (i < n) dst[i] = (_Float16)0.0f;
}

__global__ void copy_f32(const float* __restrict__ src,
                         float* __restrict__ dst, int n) {
  int i = blockIdx.x * blockDim.x + threadIdx.x;
  if (i < n) dst[i] = src[i];
}

// ---------- one recurrence step ----------
// D(128x512) = relu(H_prev)(128x512) @ W_hid^T + X_s(128x128) @ W_in^T
// h_new = 0.5*h_prev + 0.5*(D + noise_s);  out[s] = h_new;  stage relu(h_new) f16
__launch_bounds__(256)
__global__ void rnn_step(int s,
                         const _Float16* __restrict__ Xf,    // (S*B, I) f16
                         const _Float16* __restrict__ Whf,   // (H, H)  f16, [g][h]
                         const _Float16* __restrict__ Wif,   // (H, I)  f16, [g][i]
                         const _Float16* __restrict__ Hin,   // (B, H)  f16 relu(prev)
                         _Float16* __restrict__ Hout,        // (B, H)  f16 relu(new)
                         const float* __restrict__ prevH,    // (B, H)  f32 = out[s-1]
                         float* __restrict__ outS,           // (B, H)  f32 = out[s]
                         const float* __restrict__ noiseS) { // (H,)
  // Padded LDS (row pitch 520/136 halfwords) -> bank-conflict-free b128 reads
  __shared__ __align__(16) _Float16 lwh[16 * 520];
  __shared__ __align__(16) _Float16 lwi[16 * 136];

  const int tid   = threadIdx.x;
  const int gBase = blockIdx.x * 16;  // N-tile base == W row base

  // Stage W_hid slab (16 x 512 f16): 1024 x 16B chunks, 4 per thread
#pragma unroll
  for (int j = 0; j < 4; ++j) {
    int c  = tid + 256 * j;
    int r  = c >> 6;            // 64 chunks per row
    int k8 = (c & 63) << 3;
    *(f16x8*)&lwh[r * 520 + k8] =
        *(const f16x8*)&Whf[(gBase + r) * HIDDEN_ + k8];
  }
  // Stage W_in slab (16 x 128 f16): 256 chunks, 1 per thread
  {
    int r  = tid >> 4;
    int k8 = (tid & 15) << 3;
    *(f16x8*)&lwi[r * 136 + k8] =
        *(const f16x8*)&Wif[(gBase + r) * INPUT_ + k8];
  }
  __syncthreads();

  const int wave = tid >> 5;
  const int lane = tid & 31;
  const int l16  = lane & 15;
  const int half = lane >> 4;
  const int rowA = wave * 16 + l16;  // batch row held by this lane (A frag: M = lane%16)
  const int koff = half * 8;         // ISA A/B frag K slicing: {8h..8h+7, 16+8h..16+8h+7}

  v8f acc = {0.f, 0.f, 0.f, 0.f, 0.f, 0.f, 0.f, 0.f};

  // Hidden contribution: K = 512 (16 WMMA k-steps)
  {
    const _Float16* aRow = Hin + rowA * HIDDEN_;
    const _Float16* bRow = &lwh[l16 * 520];
#pragma unroll
    for (int kb = 0; kb < HIDDEN_; kb += 32) {
      f16x8 a0 = *(const f16x8*)(aRow + kb + koff);
      f16x8 a1 = *(const f16x8*)(aRow + kb + koff + 16);
      f16x8 b0 = *(const f16x8*)(bRow + kb + koff);
      f16x8 b1 = *(const f16x8*)(bRow + kb + koff + 16);
      v16h a = __builtin_shufflevector(a0, a1, 0,1,2,3,4,5,6,7,8,9,10,11,12,13,14,15);
      v16h b = __builtin_shufflevector(b0, b1, 0,1,2,3,4,5,6,7,8,9,10,11,12,13,14,15);
      acc = __builtin_amdgcn_wmma_f32_16x16x32_f16(false, a, false, b,
                                                   (short)0, acc, false, false);
    }
  }
  // Input-projection contribution: K = 128 (4 WMMA k-steps)
  {
    const _Float16* xRow = Xf + (size_t)(s * BATCH_ + rowA) * INPUT_;
    const _Float16* bRow = &lwi[l16 * 136];
#pragma unroll
    for (int kb = 0; kb < INPUT_; kb += 32) {
      f16x8 a0 = *(const f16x8*)(xRow + kb + koff);
      f16x8 a1 = *(const f16x8*)(xRow + kb + koff + 16);
      f16x8 b0 = *(const f16x8*)(bRow + kb + koff);
      f16x8 b1 = *(const f16x8*)(bRow + kb + koff + 16);
      v16h a = __builtin_shufflevector(a0, a1, 0,1,2,3,4,5,6,7,8,9,10,11,12,13,14,15);
      v16h b = __builtin_shufflevector(b0, b1, 0,1,2,3,4,5,6,7,8,9,10,11,12,13,14,15);
      acc = __builtin_amdgcn_wmma_f32_16x16x32_f16(false, a, false, b,
                                                   (short)0, acc, false, false);
    }
  }

  // Epilogue: leak + noise + store + relu->f16 staging for next step.
  // C/D layout: lanes 0-15: (M=v, N=lane); lanes 16-31: (M=8+v, N=lane-16)
#pragma unroll
  for (int v = 0; v < 8; ++v) {
    int row = wave * 16 + half * 8 + v;
    int col = gBase + l16;
    float hp = 0.0f;
    if (s > 0) hp = prevH[row * HIDDEN_ + col];   // uniform branch (scalar s)
    float val = 0.5f * hp + 0.5f * (acc[v] + noiseS[col]);
    outS[row * HIDDEN_ + col] = val;
    Hout[row * HIDDEN_ + col] = (_Float16)fmaxf(val, 0.0f);
  }
}

extern "C" void kernel_launch(void* const* d_in, const int* in_sizes, int n_in,
                              void* d_out, int out_size, void* d_ws, size_t ws_size,
                              hipStream_t stream) {
  const float* x     = (const float*)d_in[0];  // (S, B, I)
  const float* W_in  = (const float*)d_in[1];  // (H, I)
  const float* W_hid = (const float*)d_in[2];  // (H, H)
  const float* noise = (const float*)d_in[3];  // (S, H)
  float* out = (float*)d_out;                  // (S*B*H) trajectory + (B*H) final

  const int SBI = SEQ_ * BATCH_ * INPUT_;
  const int HH  = HIDDEN_ * HIDDEN_;
  const int HI  = HIDDEN_ * INPUT_;
  const int BH  = BATCH_ * HIDDEN_;

  // workspace layout (f16), ~17.7 MB total
  _Float16* xf   = (_Float16*)d_ws;
  _Float16* whf  = xf + SBI;
  _Float16* wif  = whf + HH;
  _Float16* hst0 = wif + HI;
  _Float16* hst1 = hst0 + BH;

  cvt_f32_to_f16<<<(SBI + 255) / 256, 256, 0, stream>>>(x, xf, SBI);
  cvt_f32_to_f16<<<(HH + 255) / 256, 256, 0, stream>>>(W_hid, whf, HH);
  cvt_f32_to_f16<<<(HI + 255) / 256, 256, 0, stream>>>(W_in, wif, HI);
  zero_f16<<<(BH + 255) / 256, 256, 0, stream>>>(hst0, BH);

  for (int s = 0; s < SEQ_; ++s) {
    const _Float16* hin = (s & 1) ? hst1 : hst0;
    _Float16* hout      = (s & 1) ? hst0 : hst1;
    const float* prevH  = out + (size_t)(s > 0 ? s - 1 : 0) * BH;
    rnn_step<<<dim3(HIDDEN_ / 16), 256, 0, stream>>>(
        s, xf, whf, wif, hin, hout, prevH,
        out + (size_t)s * BH, noise + (size_t)s * HIDDEN_);
  }

  // final hidden state = out[S-1], appended after the trajectory
  copy_f32<<<(BH + 255) / 256, 256, 0, stream>>>(
      out + (size_t)(SEQ_ - 1) * BH, out + (size_t)SEQ_ * BH, BH);
}
